// MaskAttention_25374666785047
// MI455X (gfx1250) — compile-verified, matrix operand order checked by reading, not run on hardware
//
#include <hip/hip_runtime.h>
#include <cstdint>

#define B_TOTAL 65536
#define LSEQ 9
#define DIM 128
#define ODIM 64
#define TB 16
#define NWAVES 4
#define W1_STRIDE 272   // 256 + 16 bf16 pad -> 544B row stride, 32B aligned, conflict-light
#define FCW_STRIDE 144  // 128 + 16 bf16 pad -> 288B row stride, 32B aligned

typedef __attribute__((ext_vector_type(16))) __bf16 v16bf;
typedef __attribute__((ext_vector_type(8)))  float  v8f;

struct __align__(32) Smem {
  __bf16 w1[DIM * W1_STRIDE];     // [n=0..127][k2=0..255] (k2<128: key part, >=128: value part)
  __bf16 fcw[ODIM * FCW_STRIDE];  // [n=0..63][k=0..127]
  float  red[NWAVES][2][8][16];   // per-wave logit transpose scratch
  float  attn[NWAVES][16][12];    // per-wave logits -> attention probs
};

static __device__ inline v8f zero8() {
  v8f z;
  #pragma unroll
  for (int i = 0; i < 8; ++i) z[i] = 0.0f;
  return z;
}

// load 16 contiguous f32 for this lane and convert to bf16 (A-operand half-row)
static __device__ inline v16bf cvt_row16(const float* __restrict__ p) {
  const float4* q = (const float4*)p;
  v16bf r;
  #pragma unroll
  for (int i = 0; i < 4; ++i) {
    float4 f = q[i];
    r[4*i+0] = (__bf16)f.x; r[4*i+1] = (__bf16)f.y;
    r[4*i+2] = (__bf16)f.z; r[4*i+3] = (__bf16)f.w;
  }
  return r;
}

// D = A*B + C, B-column (16 bf16) read from LDS
static __device__ inline v8f wmma_b(v16bf a, const __bf16* bp, v8f c) {
  v16bf b = *(const v16bf*)bp;
  return __builtin_amdgcn_wmma_f32_16x16x32_bf16(false, a, false, b, (short)0, c,
                                                 false, false);
}

// same-wave LDS write -> cross-lane read ordering (DS is in-order per wave;
// the wait + memory clobber stops the compiler from reordering around it)
static __device__ inline void lds_fence() {
  asm volatile("s_wait_dscnt 0x0" ::: "memory");
}

__global__ __launch_bounds__(NWAVES * 32, 1) void
mask_attn_kernel(const float* __restrict__ value,
                 const float* __restrict__ key,
                 const int*   __restrict__ seq_len,
                 const float* __restrict__ w1,
                 const float* __restrict__ b1,
                 const float* __restrict__ w2g,
                 const float* __restrict__ b2g,
                 const float* __restrict__ fcw,
                 const float* __restrict__ fcb,
                 float* __restrict__ out)
{
  __shared__ Smem sm;
  const int tid = threadIdx.x;

  // ---- cooperative weight preload: f32 -> bf16 into LDS ----
  for (int i = tid; i < DIM * 2 * DIM; i += NWAVES * 32) {
    int n = i >> 8, k = i & 255;
    sm.w1[n * W1_STRIDE + k] = (__bf16)w1[i];
  }
  for (int i = tid; i < ODIM * DIM; i += NWAVES * 32) {
    int n = i >> 7, k = i & 127;
    sm.fcw[n * FCW_STRIDE + k] = (__bf16)fcw[i];
  }
  __syncthreads();

  const int wave = tid >> 5;
  const int lane = tid & 31;
  const int lm = lane & 15;   // A: row in tile / B,C: column (low 4 bits)
  const int lh = lane >> 4;   // half: A: K-half / C: row group
  const int b0 = (blockIdx.x * NWAVES + wave) * TB;
  const int bm = b0 + lm;     // this lane's batch row

  // per-lane column constants (column n = t*16 + lm)
  float b1v[8], w2v[8];
  #pragma unroll
  for (int t = 0; t < 8; ++t) {
    b1v[t] = b1[t * 16 + lm];
    w2v[t] = w2g[t * 16 + lm];
  }
  float fcbv[4];
  #pragma unroll
  for (int t = 0; t < 4; ++t) fcbv[t] = fcb[t * 16 + lm];
  const float b2v = b2g[0];

  // sequence length for this row; tile-wide max bounds the l loop
  int seq = seq_len[bm];
  seq = min(max(seq, 0), LSEQ - 1);
  int smax = seq;
  #pragma unroll
  for (int d = 1; d < 16; d <<= 1) {
    int o = __shfl_xor(smax, d, 32);
    smax = max(smax, o);
  }
  const int lcnt = smax + 1;  // valid positions are 0..smax (t <= seq_len)

  // ---- key partial: kp = key_tile @ w1a^T, shared across all positions ----
  v8f kp[8];
  #pragma unroll
  for (int t = 0; t < 8; ++t) kp[t] = zero8();
  {
    const float* kr = key + (size_t)bm * DIM + lh * 16;
    #pragma unroll
    for (int kc = 0; kc < 4; ++kc) {
      v16bf a = cvt_row16(kr + kc * 32);
      #pragma unroll
      for (int t = 0; t < 8; ++t)
        kp[t] = wmma_b(a, &sm.w1[(t * 16 + lm) * W1_STRIDE + kc * 32 + lh * 16], kp[t]);
    }
  }

  // ---- per-position logits: h = relu(kp + value_l @ w1b^T + b1); logit = h.w2 + b2
  const float* vrow = value + (size_t)bm * LSEQ * DIM + lh * 16;
  for (int l = 0; l < lcnt; ++l) {
    v8f h[8];
    #pragma unroll
    for (int t = 0; t < 8; ++t) h[t] = kp[t];
    #pragma unroll
    for (int kc = 0; kc < 4; ++kc) {
      v16bf a = cvt_row16(vrow + (size_t)l * DIM + kc * 32);
      #pragma unroll
      for (int t = 0; t < 8; ++t)
        h[t] = wmma_b(a, &sm.w1[(t * 16 + lm) * W1_STRIDE + DIM + kc * 32 + lh * 16], h[t]);
    }
    // relu + scale by w2, partial row sums over this lane's 8 columns
    float part[8];
    #pragma unroll
    for (int r = 0; r < 8; ++r) part[r] = 0.0f;
    #pragma unroll
    for (int t = 0; t < 8; ++t) {
      #pragma unroll
      for (int r = 0; r < 8; ++r) {
        float x = h[t][r] + b1v[t];
        x = fmaxf(x, 0.0f);
        part[r] = fmaf(x, w2v[t], part[r]);
      }
    }
    // transpose-reduce across the 16 column-lanes via LDS
    #pragma unroll
    for (int r = 0; r < 8; ++r) sm.red[wave][lh][r][lm] = part[r];
    lds_fence();
    const float4* rp = (const float4*)&sm.red[wave][lm >> 3][lm & 7][0];
    float4 s0 = rp[0], s1 = rp[1], s2 = rp[2], s3 = rp[3];
    float logit = b2v +
        (s0.x + s0.y + s0.z + s0.w) + (s1.x + s1.y + s1.z + s1.w) +
        (s2.x + s2.y + s2.z + s2.w) + (s3.x + s3.y + s3.z + s3.w);
    if (lane < 16) sm.attn[wave][lm][l] = logit;
    lds_fence();
  }

  // ---- masked softmax (lane owns row lm; both halves compute identically) ----
  float e[LSEQ];
  float mx = -3.0e38f;
  #pragma unroll
  for (int l = 0; l < LSEQ; ++l)
    if (l <= seq) mx = fmaxf(mx, sm.attn[wave][lm][l]);
  float den = 0.0f;
  #pragma unroll
  for (int l = 0; l < LSEQ; ++l) {
    float v = 0.0f;
    if (l <= seq) { v = __expf(sm.attn[wave][lm][l] - mx); den += v; }
    e[l] = v;
  }
  float inv = 1.0f / den;
  if (lane < 16) {
    #pragma unroll
    for (int l = 0; l < LSEQ; ++l) sm.attn[wave][lm][l] = e[l] * inv;
  }
  lds_fence();

  // ---- out = attn-weighted sum of value; then fc (bf16 WMMA) + relu ----
  v8f acc[4];
  #pragma unroll
  for (int t = 0; t < 4; ++t) acc[t] = zero8();

  #pragma unroll
  for (int j = 0; j < 4; ++j) {           // K-chunk of the fc GEMM
    float oc[16];
    #pragma unroll
    for (int i = 0; i < 16; ++i) oc[i] = 0.0f;
    for (int l = 0; l < lcnt; ++l) {
      float a = sm.attn[wave][lm][l];
      if (a != 0.0f) {                    // lane-divergent skip; EXEC restored after
        const float4* q = (const float4*)(vrow + (size_t)l * DIM + j * 32);
        #pragma unroll
        for (int i2 = 0; i2 < 4; ++i2) {
          float4 f = q[i2];
          oc[4*i2+0] = fmaf(a, f.x, oc[4*i2+0]);
          oc[4*i2+1] = fmaf(a, f.y, oc[4*i2+1]);
          oc[4*i2+2] = fmaf(a, f.z, oc[4*i2+2]);
          oc[4*i2+3] = fmaf(a, f.w, oc[4*i2+3]);
        }
      }
    }
    v16bf a16;
    #pragma unroll
    for (int i = 0; i < 16; ++i) a16[i] = (__bf16)oc[i];
    #pragma unroll
    for (int t = 0; t < 4; ++t)
      acc[t] = wmma_b(a16, &sm.fcw[(t * 16 + lm) * FCW_STRIDE + j * 32 + lh * 16], acc[t]);
  }

  // ---- bias + relu + store (C layout: col n = t*16+lm, row m = lh*8+r) ----
  #pragma unroll
  for (int t = 0; t < 4; ++t) {
    #pragma unroll
    for (int r = 0; r < 8; ++r) {
      float x = acc[t][r] + fcbv[t];
      x = fmaxf(x, 0.0f);
      out[(size_t)(b0 + lh * 8 + r) * ODIM + t * 16 + lm] = x;
    }
  }
}

extern "C" void kernel_launch(void* const* d_in, const int* in_sizes, int n_in,
                              void* d_out, int out_size, void* d_ws, size_t ws_size,
                              hipStream_t stream) {
  (void)in_sizes; (void)n_in; (void)out_size; (void)d_ws; (void)ws_size;
  const float* value = (const float*)d_in[0];
  const float* key   = (const float*)d_in[1];
  const int*   seq   = (const int*)d_in[2];
  // d_in[3] = maxlen (compile-time 9)
  const float* w1  = (const float*)d_in[4];
  const float* b1  = (const float*)d_in[5];
  const float* w2  = (const float*)d_in[6];
  const float* b2  = (const float*)d_in[7];
  const float* fcw = (const float*)d_in[8];
  const float* fcb = (const float*)d_in[9];
  float* out = (float*)d_out;

  dim3 grid(B_TOTAL / (NWAVES * TB));   // 1024 blocks
  dim3 block(NWAVES * 32);              // 4 waves of 32
  hipLaunchKernelGGL(mask_attn_kernel, grid, block, 0, stream,
                     value, key, seq, w1, b1, w2, b2, fcw, fcb, out);
}
